// AFP_71116068487938
// MI455X (gfx1250) — matrix operand order repaired and checked by reading, not compiled, not run on hardware
//
#include <hip/hip_runtime.h>

// ---------------- constants from the reference ----------------
#define BB     4
#define PIX_C  256
#define PE_C   64
#define LAT_C  256
#define NS_C   128
#define NH_C   4
#define HD_C   64
#define ITERS_C 3
#define NKMAX  32768   // res0: 128*256

typedef __attribute__((ext_vector_type(16))) __bf16 v16bf;
typedef __attribute__((ext_vector_type(8)))  float  v8f;

union FragU { v16bf v; uint4 q[2]; };

__device__ inline unsigned short f2bf(float f) {
  unsigned u = __float_as_uint(f);
  u += 0x7FFFu + ((u >> 16) & 1u);        // round-to-nearest-even
  return (unsigned short)(u >> 16);
}
__device__ inline unsigned f2ord(float f) { // monotone f32 -> u32 map
  unsigned u = __float_as_uint(f);
  return (u & 0x80000000u) ? ~u : (u | 0x80000000u);
}
__device__ inline float ord2f(unsigned u) {
  unsigned x = (u & 0x80000000u) ? (u & 0x7FFFFFFFu) : ~u;
  return __uint_as_float(x);
}
__device__ inline float wave_sum(float v) {
  #pragma unroll
  for (int m = 16; m >= 1; m >>= 1) v += __shfl_xor(v, m, 32);
  return v;
}

// ---- WMMA fragment loads (row-major A, row-major B-transposed) ----
// A 16x32 bf16: lane<16 holds row M=lane, K = {k0..k0+7, k0+16..k0+23};
//               lane>=16 holds row M=lane-16, K shifted by +8.
__device__ inline v16bf load_a_frag(const unsigned short* A, int lda, int mBase, int k0, int lane) {
  int m    = mBase + (lane & 15);
  int koff = k0 + ((lane >> 4) << 3);
  const uint4* p = reinterpret_cast<const uint4*>(A + (long long)m * lda + koff);
  FragU u; u.q[0] = p[0]; u.q[1] = p[2];  // +16 elems = +2 uint4
  return u.v;
}
// B 32x16 bf16 from Bt (N x K): lane n<16 -> column n, K = {k0..k0+7, k0+16..k0+23};
//                               lane>=16 -> column lane-16, K shifted by +8.
__device__ inline v16bf load_b_frag(const unsigned short* Bt, int ldb, int nBase, int k0, int lane) {
  int n    = nBase + (lane & 15);
  int koff = k0 + ((lane >> 4) << 3);
  const uint4* p = reinterpret_cast<const uint4*>(Bt + (long long)n * ldb + koff);
  FragU u; u.q[0] = p[0]; u.q[1] = p[2];
  return u.v;
}

// ---------------- generic bf16 GEMM: C = A @ Bt^T ----------------
// block = 256 thr = 8 waves; tile 128(M) x 64(N); z = zb*nh + zh (batch/head).
// k-loop is software-pipelined: next A + 4 B fragments are in flight while the
// current 4 WMMAs execute, so s_wait_loadcnt overlaps matrix math.
// MODE: 0 bf16 row-major (+bias)   1 bf16 transposed (+bias)
//       2 f32 store (scaled)       3 f32 accumulate into Out (+bias)
//       4 bf16 silu(x+bias)
template<int MODE>
__global__ __launch_bounds__(256) void gemm_kernel(
    const unsigned short* __restrict__ A,  long long aZb, long long aZh, int lda,
    const unsigned short* __restrict__ Bt, long long bZb, long long bZh, int ldb,
    const float* __restrict__ bias,
    void* __restrict__ OutV, long long oZb, long long oZh, int ldo,
    int Kd, int nh, float scale)
{
  const int wave  = threadIdx.x >> 5;
  const int lane  = threadIdx.x & 31;
  const int mBase = blockIdx.x * 128 + wave * 16;
  const int nBase = blockIdx.y * 64;
  const int z  = blockIdx.z;
  const int zb = z / nh, zh = z % nh;
  A  += (long long)zb * aZb + (long long)zh * aZh;
  Bt += (long long)zb * bZb + (long long)zh * bZh;
  const long long oOff = (long long)zb * oZb + (long long)zh * oZh;

  v8f zero8 = {0.f, 0.f, 0.f, 0.f, 0.f, 0.f, 0.f, 0.f};
  v8f acc[4] = {zero8, zero8, zero8, zero8};

  // prologue: fragments for kt = 0
  v16bf aC  = load_a_frag(A, lda, mBase, 0, lane);
  v16bf bC0 = load_b_frag(Bt, ldb, nBase +  0, 0, lane);
  v16bf bC1 = load_b_frag(Bt, ldb, nBase + 16, 0, lane);
  v16bf bC2 = load_b_frag(Bt, ldb, nBase + 32, 0, lane);
  v16bf bC3 = load_b_frag(Bt, ldb, nBase + 48, 0, lane);

  int kt = 0;
  for (; kt + 32 < Kd; kt += 32) {
    const int kn = kt + 32;
    if (kn + 32 < Kd)  // global_prefetch_b8, WGP scope (locality 3)
      __builtin_prefetch(A + (long long)(mBase + (lane & 15)) * lda + kn + 32, 0, 3);
    v16bf aN  = load_a_frag(A, lda, mBase, kn, lane);
    v16bf bN0 = load_b_frag(Bt, ldb, nBase +  0, kn, lane);
    v16bf bN1 = load_b_frag(Bt, ldb, nBase + 16, kn, lane);
    v16bf bN2 = load_b_frag(Bt, ldb, nBase + 32, kn, lane);
    v16bf bN3 = load_b_frag(Bt, ldb, nBase + 48, kn, lane);
    acc[0] = __builtin_amdgcn_wmma_f32_16x16x32_bf16(false, aC, false, bC0, (short)0, acc[0], false, false);
    acc[1] = __builtin_amdgcn_wmma_f32_16x16x32_bf16(false, aC, false, bC1, (short)0, acc[1], false, false);
    acc[2] = __builtin_amdgcn_wmma_f32_16x16x32_bf16(false, aC, false, bC2, (short)0, acc[2], false, false);
    acc[3] = __builtin_amdgcn_wmma_f32_16x16x32_bf16(false, aC, false, bC3, (short)0, acc[3], false, false);
    aC = aN; bC0 = bN0; bC1 = bN1; bC2 = bN2; bC3 = bN3;
  }
  // epilogue k-step
  acc[0] = __builtin_amdgcn_wmma_f32_16x16x32_bf16(false, aC, false, bC0, (short)0, acc[0], false, false);
  acc[1] = __builtin_amdgcn_wmma_f32_16x16x32_bf16(false, aC, false, bC1, (short)0, acc[1], false, false);
  acc[2] = __builtin_amdgcn_wmma_f32_16x16x32_bf16(false, aC, false, bC2, (short)0, acc[2], false, false);
  acc[3] = __builtin_amdgcn_wmma_f32_16x16x32_bf16(false, aC, false, bC3, (short)0, acc[3], false, false);

  // C/D layout: VGPR r, lane: M = mBase + r + 8*(lane>=16), N = nBase + nt*16 + (lane&15)
  const int hi = lane >> 4, ln = lane & 15;
  #pragma unroll
  for (int nt = 0; nt < 4; nt++) {
    #pragma unroll
    for (int r = 0; r < 8; r++) {
      int m = mBase + r + 8 * hi;
      int n = nBase + nt * 16 + ln;
      float val = acc[nt][r] * scale;
      if (bias) val += bias[n];
      if (MODE == 0) {
        ((unsigned short*)OutV)[oOff + (long long)m * ldo + n] = f2bf(val);
      } else if (MODE == 1) {
        ((unsigned short*)OutV)[oOff + (long long)n * ldo + m] = f2bf(val);
      } else if (MODE == 2) {
        ((float*)OutV)[oOff + (long long)m * ldo + n] = val;
      } else if (MODE == 3) {
        float* p = (float*)OutV + oOff + (long long)m * ldo + n;
        *p += val;
      } else { // MODE 4: silu
        float sv = val / (1.0f + __expf(-val));
        ((unsigned short*)OutV)[oOff + (long long)m * ldo + n] = f2bf(sv);
      }
    }
  }
}

// -------- fused sine-PE + key LayerNorm -> kin bf16 (b*nk, 320) --------
// fm layout (B, 256, H, W); one wave per key.
__global__ __launch_bounds__(256) void pe_kin_kernel(
    const float* __restrict__ fm, const float* __restrict__ lat, const float* __restrict__ lon,
    const float* __restrict__ lnk_w, const float* __restrict__ lnk_b,
    unsigned short* __restrict__ kin, int H, int Wd)
{
  const int nk   = H * Wd;
  const int wid  = blockIdx.x * 8 + (threadIdx.x >> 5);
  const int lane = threadIdx.x & 31;
  const int b = wid / nk, key = wid % nk;
  const int y = key / Wd, xw = key % Wd;

  const float* xcol = fm + (long long)b * 256 * nk + key;
  float xv[8]; float s = 0.f;
  #pragma unroll
  for (int j = 0; j < 8; j++) { xv[j] = xcol[(long long)(lane + 32 * j) * nk]; s += xv[j]; }
  float mean = wave_sum(s) * (1.0f / 256.0f);
  float d2 = 0.f;
  #pragma unroll
  for (int j = 0; j < 8; j++) { float d = xv[j] - mean; d2 += d * d; }
  float inv = rsqrtf(wave_sum(d2) * (1.0f / 256.0f) + 1e-6f);

  unsigned short* row = kin + (long long)wid * 320;
  #pragma unroll
  for (int j = 0; j < 8; j++) {
    int c = lane + 32 * j;
    row[c] = f2bf((xv[j] - mean) * inv * lnk_w[c] + lnk_b[c]);
  }
  // sine PE: pe[0:32] from lat(y), pe[32:64] from long(x); pairs (sin, cos)
  float la0 = lat[b * 2], la1 = lat[b * 2 + 1];
  float lo0 = lon[b * 2], lo1 = lon[b * 2 + 1];
  float yv = la0 + (la1 - la0) * ((y  + 0.5f) / (float)H);
  float xp = lo0 + (lo1 - lo0) * ((xw + 0.5f) / (float)Wd);
  int j0 = 2 * lane;
  float p = (j0 < 32) ? yv : xp;
  int   i = (j0 & 31) >> 1;
  float t = __powf(10000.0f, (float)i * (1.0f / 16.0f));
  float ang = p / t;
  row[256 + j0]     = f2bf(__sinf(ang));
  row[256 + j0 + 1] = f2bf(__cosf(ang));
}

// -------- slot LN + PE concat -> qin bf16 (b*128, 320); one wave/row --------
__global__ __launch_bounds__(256) void qin_kernel(
    const float* __restrict__ idr, const float* __restrict__ w, const float* __restrict__ bia,
    unsigned short* __restrict__ qin)
{
  const int wid  = blockIdx.x * 8 + (threadIdx.x >> 5);
  const int lane = threadIdx.x & 31;
  const float* row = idr + (long long)wid * 320;
  float xv[8]; float s = 0.f;
  #pragma unroll
  for (int j = 0; j < 8; j++) { xv[j] = row[lane + 32 * j]; s += xv[j]; }
  float mean = wave_sum(s) * (1.0f / 256.0f);
  float d2 = 0.f;
  #pragma unroll
  for (int j = 0; j < 8; j++) { float d = xv[j] - mean; d2 += d * d; }
  float inv = rsqrtf(wave_sum(d2) * (1.0f / 256.0f) + 1e-6f);
  unsigned short* o = qin + (long long)wid * 320;
  #pragma unroll
  for (int j = 0; j < 8; j++) {
    int c = lane + 32 * j;
    o[c] = f2bf((xv[j] - mean) * inv * w[c] + bia[c]);
  }
  int j0 = 2 * lane;
  o[256 + j0]     = f2bf(row[256 + j0]);
  o[257 + j0]     = f2bf(row[257 + j0]);
}

// -------- LN over idr x-part -> lnc bf16 (b*128, 256) --------
__global__ __launch_bounds__(256) void ln2_kernel(
    const float* __restrict__ idr, const float* __restrict__ w, const float* __restrict__ bia,
    unsigned short* __restrict__ dst)
{
  const int wid  = blockIdx.x * 8 + (threadIdx.x >> 5);
  const int lane = threadIdx.x & 31;
  const float* row = idr + (long long)wid * 320;
  float xv[8]; float s = 0.f;
  #pragma unroll
  for (int j = 0; j < 8; j++) { xv[j] = row[lane + 32 * j]; s += xv[j]; }
  float mean = wave_sum(s) * (1.0f / 256.0f);
  float d2 = 0.f;
  #pragma unroll
  for (int j = 0; j < 8; j++) { float d = xv[j] - mean; d2 += d * d; }
  float inv = rsqrtf(wave_sum(d2) * (1.0f / 256.0f) + 1e-6f);
  unsigned short* o = dst + (long long)wid * 256;
  #pragma unroll
  for (int j = 0; j < 8; j++) {
    int c = lane + 32 * j;
    o[c] = f2bf((xv[j] - mean) * inv * w[c] + bia[c]);
  }
}

// -------- exact top-k threshold (3-pass radix select) + softmax -> P bf16 ----
// one block per (b, h, q) row of nk scores. Digits: 11 / 11 / 10 bits over
// order-mapped f32 keys; 2048 u32 LDS bins. Row max is fused into pass 0,
// so the f32 score row is touched 5x total (3 select + 1 sum + 1 write).
__global__ __launch_bounds__(256) void topk_softmax_kernel(
    const float* __restrict__ S, unsigned short* __restrict__ P, int nk, int topk)
{
  const long long row = blockIdx.x;
  const float* s = S + row * (long long)nk;
  unsigned short* p = P + row * (long long)nk;
  __shared__ unsigned bins[2048];
  __shared__ unsigned s_prefix, s_remain;
  __shared__ float red[256];
  const int t = threadIdx.x;
  if (t == 0) { s_prefix = 0u; s_remain = (unsigned)topk; }
  __syncthreads();

  float lm = -3.4e38f;                       // row max, fused into pass 0
  const int shifts[3] = {21, 10, 0};
  const int nbcount[3] = {2048, 2048, 1024};
  #pragma unroll
  for (int pass = 0; pass < 3; pass++) {
    const int shift = shifts[pass];
    const int nb    = nbcount[pass];
    for (int i = t; i < nb; i += 256) bins[i] = 0u;
    __syncthreads();
    const unsigned pref = s_prefix;
    const unsigned mhi  = (pass == 0) ? 0u : (0xFFFFFFFFu << (shifts[pass - 1]));
    for (int j = t; j < nk; j += 256) {
      float v = s[j];
      if (pass == 0) lm = fmaxf(lm, v);
      unsigned u = f2ord(v);
      if ((u & mhi) == pref) atomicAdd(&bins[(u >> shift) & (nb - 1)], 1u);
    }
    __syncthreads();
    if (t == 0) {
      unsigned rem = s_remain;
      int d = nb - 1;
      while (d > 0 && bins[d] < rem) { rem -= bins[d]; d--; }
      s_prefix = pref | ((unsigned)d << shift);
      s_remain = rem;
    }
    __syncthreads();
  }
  const float thr = ord2f(s_prefix);         // kth largest; keep s >= thr (ties kept)

  // reduce row max (max is always kept since max >= thr)
  red[t] = lm; __syncthreads();
  for (int st = 128; st > 0; st >>= 1) { if (t < st) red[t] = fmaxf(red[t], red[t + st]); __syncthreads(); }
  const float mx = red[0];
  __syncthreads();
  float ls = 0.f;
  for (int j = t; j < nk; j += 256) { float v = s[j]; if (v >= thr) ls += __expf(v - mx); }
  red[t] = ls; __syncthreads();
  for (int st = 128; st > 0; st >>= 1) { if (t < st) red[t] += red[t + st]; __syncthreads(); }
  const float invs = 1.0f / red[0];
  for (int j = t; j < nk; j += 256) {
    float v = s[j];
    p[j] = (v >= thr) ? f2bf(__expf(v - mx) * invs) : (unsigned short)0;
  }
}

// -------- misc small kernels --------
__global__ __launch_bounds__(256) void transpose_w_kernel(
    const float* __restrict__ w, unsigned short* __restrict__ wT, int K, int N)
{
  int idx = blockIdx.x * 256 + threadIdx.x;     // over N*K, wT is (N, K)
  if (idx >= K * N) return;
  int n = idx / K, k = idx % K;
  wT[idx] = f2bf(w[(long long)k * N + n]);
}
__global__ __launch_bounds__(256) void init_idr_kernel(const float* __restrict__ mu, float* __restrict__ idr)
{
  int i = blockIdx.x * 256 + threadIdx.x;       // B*128*320 total
  idr[i] = mu[i % (NS_C * 320)];
}
__global__ __launch_bounds__(256) void copy_out_kernel(const float* __restrict__ src, float* __restrict__ dst)
{
  int i = blockIdx.x * 256 + threadIdx.x;
  dst[i] = src[i];
}

// ======================= host orchestration =======================
extern "C" void kernel_launch(void* const* d_in, const int* in_sizes, int n_in,
                              void* d_out, int out_size, void* d_ws, size_t ws_size,
                              hipStream_t stream)
{
  (void)in_sizes; (void)n_in; (void)out_size; (void)ws_size;
  const float* fms[2] = { (const float*)d_in[0], (const float*)d_in[1] };
  const float* lat = (const float*)d_in[2];
  const float* lon = (const float*)d_in[3];

  // params pytree leaves, sorted keys per res:
  // b1,b2,bk,bo,bq,bv,ln2_b,ln2_w,lnk_b,lnk_w,lns_b,lns_w,mu,w1,w2,wk,wo,wq,wv
  struct RP { const float *b1,*b2,*bk,*bo,*bq,*bv,*ln2_b,*ln2_w,*lnk_b,*lnk_w,
                          *lns_b,*lns_w,*mu,*w1,*w2,*wk,*wo,*wq,*wv; } rp[2];
  for (int r = 0; r < 2; r++) {
    const int base = 4 + 19 * r;
    rp[r].b1   = (const float*)d_in[base + 0];
    rp[r].b2   = (const float*)d_in[base + 1];
    rp[r].bk   = (const float*)d_in[base + 2];
    rp[r].bo   = (const float*)d_in[base + 3];
    rp[r].bq   = (const float*)d_in[base + 4];
    rp[r].bv   = (const float*)d_in[base + 5];
    rp[r].ln2_b= (const float*)d_in[base + 6];
    rp[r].ln2_w= (const float*)d_in[base + 7];
    rp[r].lnk_b= (const float*)d_in[base + 8];
    rp[r].lnk_w= (const float*)d_in[base + 9];
    rp[r].lns_b= (const float*)d_in[base +10];
    rp[r].lns_w= (const float*)d_in[base +11];
    rp[r].mu   = (const float*)d_in[base +12];
    rp[r].w1   = (const float*)d_in[base +13];
    rp[r].w2   = (const float*)d_in[base +14];
    rp[r].wk   = (const float*)d_in[base +15];
    rp[r].wo   = (const float*)d_in[base +16];
    rp[r].wq   = (const float*)d_in[base +17];
    rp[r].wv   = (const float*)d_in[base +18];
  }

  // workspace carve (256B aligned bump; sized for res0, reused by res1)
  char* W = (char*)d_ws;
  size_t off = 0;
  auto bump = [&](size_t bytes) -> void* {
    void* pp = W + off;
    off = (off + bytes + 255) & ~(size_t)255;
    return pp;
  };
  unsigned short *wqT[2], *wkT[2], *wvT[2], *woT[2], *w1T[2], *w2T[2];
  for (int r = 0; r < 2; r++) {
    wqT[r] = (unsigned short*)bump((size_t)256 * 320 * 2);
    wkT[r] = (unsigned short*)bump((size_t)256 * 320 * 2);
    wvT[r] = (unsigned short*)bump((size_t)256 * 256 * 2);
    woT[r] = (unsigned short*)bump((size_t)256 * 256 * 2);
    w1T[r] = (unsigned short*)bump((size_t)512 * 256 * 2);
    w2T[r] = (unsigned short*)bump((size_t)256 * 512 * 2);
  }
  unsigned short* kin = (unsigned short*)bump((size_t)BB * NKMAX * 320 * 2);
  unsigned short* Kb  = (unsigned short*)bump((size_t)BB * NKMAX * 256 * 2);
  unsigned short* Vt  = (unsigned short*)bump((size_t)BB * 256 * NKMAX * 2);
  float*          Sb  = (float*)         bump((size_t)BB * NH_C * NS_C * NKMAX * 4);
  unsigned short* Pb  = (unsigned short*)bump((size_t)BB * NH_C * NS_C * NKMAX * 2);
  float*          idr = (float*)         bump((size_t)BB * NS_C * 320 * 4);
  unsigned short* qin = (unsigned short*)bump((size_t)BB * NS_C * 320 * 2);
  unsigned short* Qm  = (unsigned short*)bump((size_t)BB * NS_C * 256 * 2);
  unsigned short* Ob  = (unsigned short*)bump((size_t)BB * NS_C * 256 * 2);
  unsigned short* lnc = (unsigned short*)bump((size_t)BB * NS_C * 256 * 2);
  unsigned short* h1  = (unsigned short*)bump((size_t)BB * NS_C * 512 * 2);

  const int HWs[2][2] = { {128, 256}, {64, 128} };

  for (int r = 0; r < 2; r++) {
    const int H = HWs[r][0], Wd = HWs[r][1];
    const int nk = H * Wd;
    const int topk = nk / 4;

    // weights -> transposed bf16
    transpose_w_kernel<<<dim3((320*256+255)/256), 256, 0, stream>>>(rp[r].wq, wqT[r], 320, 256);
    transpose_w_kernel<<<dim3((320*256+255)/256), 256, 0, stream>>>(rp[r].wk, wkT[r], 320, 256);
    transpose_w_kernel<<<dim3((256*256+255)/256), 256, 0, stream>>>(rp[r].wv, wvT[r], 256, 256);
    transpose_w_kernel<<<dim3((256*256+255)/256), 256, 0, stream>>>(rp[r].wo, woT[r], 256, 256);
    transpose_w_kernel<<<dim3((256*512+255)/256), 256, 0, stream>>>(rp[r].w1, w1T[r], 256, 512);
    transpose_w_kernel<<<dim3((512*256+255)/256), 256, 0, stream>>>(rp[r].w2, w2T[r], 512, 256);

    // one-time per resolution: PE + key LN -> kin; K and V projections
    pe_kin_kernel<<<dim3(BB * nk / 8), 256, 0, stream>>>(
        fms[r], lat, lon, rp[r].lnk_w, rp[r].lnk_b, kin, H, Wd);
    // K = kin @ wk + bk : (nk x 320) x (320 x 256) -> bf16 row-major
    gemm_kernel<0><<<dim3(nk/128, 4, BB), 256, 0, stream>>>(
        kin, (long long)nk*320, 0, 320, wkT[r], 0, 0, 320, rp[r].bk,
        Kb, (long long)nk*256, 0, 256, 320, 1, 1.0f);
    // V = kin[:, :256] @ wv + bv -> stored transposed (b, 256, nk)
    gemm_kernel<1><<<dim3(nk/128, 4, BB), 256, 0, stream>>>(
        kin, (long long)nk*320, 0, 320, wvT[r], 0, 0, 256, rp[r].bv,
        Vt, (long long)256*nk, 0, nk, 256, 1, 1.0f);

    init_idr_kernel<<<dim3(BB * NS_C * 320 / 256), 256, 0, stream>>>(rp[r].mu, idr);

    for (int it = 0; it < ITERS_C; it++) {
      // q_in = [LN(sx), spe] -> bf16
      qin_kernel<<<dim3(BB * NS_C / 8), 256, 0, stream>>>(idr, rp[r].lns_w, rp[r].lns_b, qin);
      // Q = q_in @ wq + bq
      gemm_kernel<0><<<dim3(1, 4, BB), 256, 0, stream>>>(
          qin, (long long)NS_C*320, 0, 320, wqT[r], 0, 0, 320, rp[r].bq,
          Qm, (long long)NS_C*256, 0, 256, 320, 1, 1.0f);
      // S[b,h] = Q_h @ K_h^T / 8  (f32)
      gemm_kernel<2><<<dim3(1, nk/64, BB * NH_C), 256, 0, stream>>>(
          Qm, (long long)NS_C*256, 64, 256, Kb, (long long)nk*256, 64, 256, nullptr,
          Sb, (long long)NH_C*NS_C*nk, (long long)NS_C*nk, nk, 64, NH_C, 0.125f);
      // exact top-k threshold + softmax -> P bf16
      topk_softmax_kernel<<<dim3(BB * NH_C * NS_C), 256, 0, stream>>>(Sb, Pb, nk, topk);
      // O[b,:,h*64:] = P @ V_h   (bf16 out, head slices of a (b,128,256) buffer)
      gemm_kernel<0><<<dim3(1, 1, BB * NH_C), 256, 0, stream>>>(
          Pb, (long long)NH_C*NS_C*nk, (long long)NS_C*nk, nk,
          Vt, (long long)256*nk, (long long)64*nk, nk, nullptr,
          Ob, (long long)NS_C*256, 64, 256, nk, NH_C, 1.0f);
      // idr_x += O @ wo + bo  (residual fused into epilogue)
      gemm_kernel<3><<<dim3(1, 4, BB), 256, 0, stream>>>(
          Ob, (long long)NS_C*256, 0, 256, woT[r], 0, 0, 256, rp[r].bo,
          idr, (long long)NS_C*320, 0, 320, 256, 1, 1.0f);
      // MLP: lnc = LN(idr_x); h1 = silu(lnc@w1+b1); idr_x += h1@w2+b2
      ln2_kernel<<<dim3(BB * NS_C / 8), 256, 0, stream>>>(idr, rp[r].ln2_w, rp[r].ln2_b, lnc);
      gemm_kernel<4><<<dim3(1, 8, BB), 256, 0, stream>>>(
          lnc, (long long)NS_C*256, 0, 256, w1T[r], 0, 0, 256, rp[r].b1,
          h1, (long long)NS_C*512, 0, 512, 256, 1, 1.0f);
      gemm_kernel<3><<<dim3(1, 4, BB), 256, 0, stream>>>(
          h1, (long long)NS_C*512, 0, 512, w2T[r], 0, 0, 512, rp[r].b2,
          idr, (long long)NS_C*320, 0, 320, 512, 1, 1.0f);
    }
    copy_out_kernel<<<dim3(BB * NS_C * 320 / 256), 256, 0, stream>>>(
        idr, (float*)d_out + (size_t)r * BB * NS_C * 320);
  }
}
